// QliGRU_22771916603945
// MI455X (gfx1250) — matrix operand order repaired and checked by reading, not compiled, not run on hardware
//
#include <hip/hip_runtime.h>
#include <cmath>

// ---------------- problem constants ----------------
#define TSTEPS 1000
#define BATCH  32
#define DIN    256
#define HDIM   512
#define NWG    16          // persistent workgroups, 32 H-columns each
#define NCOL   32          // output columns per workgroup
#define KTOT   768         // DIN + HDIM fused K
#define KSTEPS (KTOT / 32) // 24 WMMA k-steps
#define LSTR   776         // padded K stride (halves) -> 16B pad, bank-spread
#define THREADS 256        // 8 wave32

typedef __attribute__((ext_vector_type(16))) _Float16 v16h;
typedef __attribute__((ext_vector_type(8)))  _Float16 v8h;
typedef __attribute__((ext_vector_type(4)))  _Float16 v4h;
typedef __attribute__((ext_vector_type(8)))  float    v8f;

union HV { v16h v; v8h h[2]; };

struct QParams {
  const float* x;          // (T,B,D)
  const float* w[2][4];    // [mat][comp]: mat0=Z -> wz_{r,i,j,k}, mat1=H -> wh_*
  const float* u[2][4];    // mat0 -> uz_*, mat1 -> uh_*
  const float* zb;         // wz_b (H)
  const float* hb;         // wh_b (H)
  float*       out;        // (T,B,H) fp32
  _Float16*    hbuf0;      // (B,H) f16 double buffer
  _Float16*    hbuf1;
  unsigned*    cnt;        // grid barrier arrive counter
  unsigned*    gen;        // grid barrier generation
};

// quaternion block layout: W[rb][cb] = sgn * comp, from
// c0=[r,-i,-j,-k] c1=[i,r,-k,j] c2=[j,k,r,-i] c3=[k,-j,i,r] (rows=rb, cols=cb)
__device__ const int   q_comp[4][4] = {{0,1,2,3},{1,0,3,2},{2,3,0,1},{3,2,1,0}};
__device__ const float q_sgn [4][4] = {{ 1.f, 1.f, 1.f, 1.f},
                                       {-1.f, 1.f, 1.f,-1.f},
                                       {-1.f,-1.f, 1.f, 1.f},
                                       {-1.f, 1.f,-1.f, 1.f}};

// Device-wide sense barrier. Every wave executes the release fence (store
// counters are per-wave), one lane per WG does the atomic handshake, and every
// wave executes the acquire fence (L0 invalidate) before re-reading peer data.
__device__ __forceinline__ void grid_barrier(unsigned* cnt, unsigned* gen) {
  __builtin_amdgcn_fence(__ATOMIC_RELEASE, "agent");   // publish h(t+1) stores
  __syncthreads();
  if (threadIdx.x == 0) {
    unsigned g = __hip_atomic_load(gen, __ATOMIC_RELAXED, __HIP_MEMORY_SCOPE_AGENT);
    unsigned a = __hip_atomic_fetch_add(cnt, 1u, __ATOMIC_ACQ_REL, __HIP_MEMORY_SCOPE_AGENT);
    if (a == NWG - 1u) {
      __hip_atomic_store(cnt, 0u, __ATOMIC_RELAXED, __HIP_MEMORY_SCOPE_AGENT);
      __hip_atomic_fetch_add(gen, 1u, __ATOMIC_RELEASE, __HIP_MEMORY_SCOPE_AGENT);
    } else {
      while (__hip_atomic_load(gen, __ATOMIC_RELAXED, __HIP_MEMORY_SCOPE_AGENT) == g)
        __builtin_amdgcn_s_sleep(1);
    }
  }
  __syncthreads();
  __builtin_amdgcn_fence(__ATOMIC_ACQUIRE, "agent");   // invalidate stale L0 lines
}

__global__ void __launch_bounds__(THREADS, 1)
qligru_persistent(QParams P) {
  extern __shared__ char smem[];
  _Float16* Wt   = (_Float16*)smem;               // [2*NCOL][LSTR] f16, [n][k], k-major
  _Float16* Alds = Wt + 2 * NCOL * LSTR;          // [BATCH][LSTR]  f16, A = [x_t ; h_t]
  float*    zs   = (float*)(Alds + BATCH * LSTR); // [BATCH][NCOL]  fp32 z pre-activation
  float*    as_  = zs + BATCH * NCOL;             // [BATCH][NCOL]  fp32 hcand pre-act

  const unsigned wg   = blockIdx.x;
  const unsigned tid  = threadIdx.x;
  const unsigned lane = tid & 31u;
  const unsigned wv   = tid >> 5;   // 0..7

  // ---- one-time: assemble this WG's f16 slice of [Wz;Uz] and [Wh;Uh] ([n][k], transposed)
  for (unsigned idx = tid; idx < 2u * NCOL * KTOT; idx += THREADS) {
    unsigned mat = idx / (NCOL * KTOT);
    unsigned rem = idx % (NCOL * KTOT);
    unsigned n   = rem / KTOT;
    unsigned k   = rem % KTOT;
    unsigned gc  = wg * NCOL + n;          // global H column
    unsigned cb  = gc >> 7, bc = gc & 127u;
    unsigned rb, off;
    const float* src;
    if (k < DIN) { rb = k >> 6;  off = (k & 63u)  * 128u + bc; src = P.w[mat][q_comp[rb][cb]]; }
    else { unsigned kk = k - DIN; rb = kk >> 7; off = (kk & 127u) * 128u + bc; src = P.u[mat][q_comp[rb][cb]]; }
    Wt[(mat * NCOL + n) * LSTR + k] = (_Float16)(q_sgn[rb][cb] * src[off]);
  }

  // per-thread fixed epilogue coordinates + bias registers
  const unsigned el0 = tid * 4u;               // 4 contiguous elements of 32x32 tile
  const unsigned em  = el0 >> 5;               // batch row
  const unsigned ec0 = el0 & 31u;              // first local column (multiple of 4)
  const unsigned egc = wg * NCOL + ec0;        // first global column
  float zb4[4], hb4[4];
  #pragma unroll
  for (int i = 0; i < 4; ++i) { zb4[i] = P.zb[egc + i]; hb4[i] = P.hb[egc + i]; }

  // wave -> one 16x16 output tile of {z | hcand}
  const unsigned mat   = wv & 1u;          // 0 = z-path, 1 = hcand-path
  const unsigned mtile = (wv >> 1) & 1u;   // batch rows 0..15 / 16..31
  const unsigned ntile = (wv >> 2) & 1u;   // cols 0..15 / 16..31 of this WG's slice
  // A fragment (16-bit 16x32): lanes 0-15 hold K[0..8)+[16..24), lanes 16-31 K[8..16)+[24..32)
  const _Float16* Abase = Alds + (mtile * 16u + (lane & 15u)) * LSTR + ((lane >> 4) * 8u);
  // B fragment (16-bit 32x16): lane n = lane&15, lanes 0-15 K[0..8)+[16..24), lanes 16-31 K[8..16)+[24..32)
  const _Float16* Bbase = Wt + (mat * NCOL + ntile * 16u + (lane & 15u)) * LSTR + ((lane >> 4) * 8u);
  float* scr = mat ? as_ : zs;

  __syncthreads();  // weights ready

  for (int t = 0; t < TSTEPS; ++t) {
    const _Float16* hsrc = (t & 1) ? P.hbuf1 : P.hbuf0;
    _Float16*       hdst = (t & 1) ? P.hbuf0 : P.hbuf1;

    // ---- stage x_t -> A[:, 0:256)  (float4 loads, 8-byte f16 packs)
    const float4* x4 = (const float4*)(P.x + (size_t)t * (BATCH * DIN));
    for (unsigned i = tid; i < (BATCH * DIN) / 4u; i += THREADS) {
      float4 v = x4[i];
      unsigned m = i >> 6, k4 = (i & 63u) * 4u;   // 64 float4 per row
      v4h h4;
      h4[0] = (_Float16)v.x; h4[1] = (_Float16)v.y;
      h4[2] = (_Float16)v.z; h4[3] = (_Float16)v.w;
      *(v4h*)(Alds + m * LSTR + k4) = h4;
    }
    if (t + 1 < TSTEPS)
      __builtin_prefetch(P.x + (size_t)(t + 1) * (BATCH * DIN) + tid * 32u, 0, 0);

    // ---- stage h(t) -> A[:, 256:768)  (b128 copies; acquire fence already ran)
    const uint4* h16 = (const uint4*)hsrc;
    for (unsigned i = tid; i < (BATCH * HDIM) / 8u; i += THREADS) {
      uint4 v = h16[i];
      unsigned m = i >> 6, c8 = (i & 63u) * 8u;   // 64 uint4 (8 halves) per row
      *(uint4*)(Alds + m * LSTR + DIN + c8) = v;
    }
    __syncthreads();

    // ---- fused GEMM: [x_t ; h] (32x768) x slice (768x32) for both gates.
    // Depth-2 software pipeline: fragments for k+1,k+2 are fetched before the
    // WMMA for k issues, so the compiler can use partial s_wait_dscnt and hide
    // LDS latency under the matrix pipe. Fully unrolled -> pure renaming.
    v8f acc = {};
    {
      HV a[3], b[3];
      auto ldfrag = [&](int s, int kk) {
        const int K0 = kk * 32;
        a[s].h[0] = *(const v8h*)(Abase + K0);
        a[s].h[1] = *(const v8h*)(Abase + K0 + 16);
        b[s].h[0] = *(const v8h*)(Bbase + K0);
        b[s].h[1] = *(const v8h*)(Bbase + K0 + 16);
      };
      ldfrag(0, 0);
      ldfrag(1, 1);
      #pragma unroll
      for (int kk = 0; kk < KSTEPS; ++kk) {
        if (kk + 2 < KSTEPS) ldfrag((kk + 2) % 3, kk + 2);
        const int s = kk % 3;
        acc = __builtin_amdgcn_wmma_f32_16x16x32_f16(false, a[s].v, false, b[s].v,
                                                     (short)0, acc, false, false);
      }
    }
    {
      // C/D layout: lane n = lane&15, rows (lane>>4)*8 + v
      const unsigned n  = lane & 15u;
      const unsigned mb = (lane >> 4) * 8u;
      #pragma unroll
      for (int vv = 0; vv < 8; ++vv)
        scr[(mtile * 16u + mb + (unsigned)vv) * NCOL + ntile * 16u + n] = acc[vv];
    }
    __syncthreads();

    // ---- gate epilogue (vectorized, fast rcp sigmoid):
    //      z = sigmoid(zs+b), hcand = relu(as+b)*keep, h' = z*h + (1-z)*hcand
    {
      float4 zv = *(const float4*)(zs  + em * NCOL + ec0);
      float4 av = *(const float4*)(as_ + em * NCOL + ec0);
      v4h    hv = *(const v4h*)(Alds + em * LSTR + DIN + egc);
      float zr[4] = {zv.x, zv.y, zv.z, zv.w};
      float ar[4] = {av.x, av.y, av.z, av.w};
      float hn[4];
      #pragma unroll
      for (int i = 0; i < 4; ++i) {
        float z  = __builtin_amdgcn_rcpf(1.f + __expf(-(zr[i] + zb4[i])));
        float hc = fmaxf(ar[i] + hb4[i], 0.f) * 0.8f;   // DROP_KEEP
        float ho = (float)hv[i];
        hn[i] = z * ho + (1.f - z) * hc;
      }
      float* op = P.out + (size_t)t * (BATCH * HDIM) + em * HDIM + egc;
      *(float4*)op = make_float4(hn[0], hn[1], hn[2], hn[3]);
      v4h hp;
      hp[0] = (_Float16)hn[0]; hp[1] = (_Float16)hn[1];
      hp[2] = (_Float16)hn[2]; hp[3] = (_Float16)hn[3];
      *(v4h*)(hdst + em * HDIM + egc) = hp;
    }
    grid_barrier(P.cnt, P.gen);   // publish h(t+1), step the whole grid
  }
}

// ---------------- host ----------------
static constexpr size_t SMEM_BYTES =
    (size_t)(2 * NCOL * LSTR + BATCH * LSTR) * 2u   // f16 weights + A staging
  + (size_t)(2 * BATCH * NCOL) * 4u;                // fp32 z/a scratch  (~153.5 KB)

extern "C" void kernel_launch(void* const* d_in, const int* in_sizes, int n_in,
                              void* d_out, int out_size, void* d_ws, size_t ws_size,
                              hipStream_t stream) {
  (void)in_sizes; (void)n_in; (void)out_size; (void)ws_size;
  QParams P;
  P.x = (const float*)d_in[0];
  for (int c = 0; c < 4; ++c) {
    P.w[0][c] = (const float*)d_in[5 + c];   // wz_{r,i,j,k}
    P.w[1][c] = (const float*)d_in[1 + c];   // wh_{r,i,j,k}
    P.u[0][c] = (const float*)d_in[13 + c];  // uz_{r,i,j,k}
    P.u[1][c] = (const float*)d_in[9 + c];   // uh_{r,i,j,k}
  }
  P.hb  = (const float*)d_in[17];  // wh_b
  P.zb  = (const float*)d_in[18];  // wz_b
  P.out = (float*)d_out;

  char* ws = (char*)d_ws;
  P.hbuf0 = (_Float16*)ws;                       // 32 KB
  P.hbuf1 = (_Float16*)(ws + 32768);             // 32 KB
  P.cnt   = (unsigned*)(ws + 65536);
  P.gen   = (unsigned*)(ws + 65536 + 64);

  // zero h0 + barrier state every call (graph-capture-safe, deterministic)
  hipMemsetAsync(d_ws, 0, 65536 + 128, stream);

  hipFuncSetAttribute(reinterpret_cast<const void*>(qligru_persistent),
                      hipFuncAttributeMaxDynamicSharedMemorySize, (int)SMEM_BYTES);
  qligru_persistent<<<dim3(NWG), dim3(THREADS), SMEM_BYTES, stream>>>(P);
}
